// ClusterMemory_81320910782927
// MI455X (gfx1250) — matrix operand order repaired and checked by reading, not compiled, not run on hardware
//
#include <hip/hip_runtime.h>
#include <math.h>

typedef __attribute__((ext_vector_type(4)))  float        v4f;
typedef __attribute__((ext_vector_type(4)))  unsigned int v4u;
typedef __attribute__((ext_vector_type(8)))  unsigned int v8u;
typedef __attribute__((ext_vector_type(8)))  float        v8f;
typedef __attribute__((ext_vector_type(16))) __bf16       v16bf;

#define BDIM 256
#define DDIM 2048
#define NDIM 65536
#define PDIM 4096
#define NCLS (PDIM + NDIM)
#define INV_TEMP 20.0f
#define EPS_SM 0.1f

#define KSTEPS (DDIM / 32)                    // 64 WMMA K-steps
#define PANEL_KS 2                            // K-steps per LDS panel
#define PANEL_BYTES (PANEL_KS * 16 * 32 * 32) // 32768 B (2 ksteps x 16 mtiles x 32 lanes x 32 B)
#define NPANELS (KSTEPS / PANEL_KS)           // 32

union Frag { v16bf v; v8u u; v4u q[2]; };

// f32 pair -> packed bf16 (round-to-nearest-even)
__device__ __forceinline__ unsigned pk_bf16(float lo, float hi) {
    unsigned a = __float_as_uint(lo);
    unsigned b = __float_as_uint(hi);
    a += 0x7FFFu + ((a >> 16) & 1u);
    b += 0x7FFFu + ((b >> 16) & 1u);
    return (a >> 16) | (b & 0xFFFF0000u);
}

// CDNA5 async copy: 16 B global -> LDS, tracked by ASYNCcnt (no VGPR data regs)
__device__ __forceinline__ void async16(unsigned ldsoff, const void* g) {
    asm volatile("global_load_async_to_lds_b128 %0, %1, off"
                 :: "v"(ldsoff), "v"((unsigned long long)(size_t)g)
                 : "memory");
}

// ---------------- Kernel 1: L2-normalize rows + write bf16 in A-fragment-swizzled layout --------
// Fragment layout (bytes): ((kstep*16 + mtile)*64 + plane*32 + lane) * 16
//   lane<16  : plane0 = K kstep*32+0..7,  plane1 = +16..23   (row = mtile*16 + lane)
//   lane>=16 : plane0 = K kstep*32+8..15, plane1 = +24..31
__global__ __launch_bounds__(256) void cm_normalize(const float* __restrict__ in,
                                                    unsigned short* __restrict__ xs) {
    const int b = blockIdx.x, tid = threadIdx.x;       // thread owns 8-elem chunk c = tid
    const v4f* row = (const v4f*)(in + (size_t)b * DDIM);
    v4f a0 = row[2 * tid], a1 = row[2 * tid + 1];      // K = 8*tid .. 8*tid+7
    float s = a0[0]*a0[0] + a0[1]*a0[1] + a0[2]*a0[2] + a0[3]*a0[3]
            + a1[0]*a1[0] + a1[1]*a1[1] + a1[2]*a1[2] + a1[3]*a1[3];
    __shared__ float red[256];
    red[tid] = s; __syncthreads();
    for (int off = 128; off > 0; off >>= 1) {
        if (tid < off) red[tid] += red[tid + off];
        __syncthreads();
    }
    const float rn = rsqrtf(red[0]);

    const int kstep = tid >> 2, sub = tid & 3;
    const int lane  = (b & 15) + ((sub & 1) << 4);
    const int mtile = b >> 4,  plane = sub >> 1;
    const size_t off_h = (size_t)((kstep * 16 + mtile) * 64 + plane * 32 + lane) * 8;
    v4u w;
    w[0] = pk_bf16(a0[0]*rn, a0[1]*rn);
    w[1] = pk_bf16(a0[2]*rn, a0[3]*rn);
    w[2] = pk_bf16(a1[0]*rn, a1[1]*rn);
    w[3] = pk_bf16(a1[2]*rn, a1[3]*rn);
    *(v4u*)(xs + off_h) = w;
}

// ---------------- Kernel 2: mem_logits = x @ features^T (bf16 WMMA, async-LDS A-panels) ---------
__global__ __launch_bounds__(256) void cm_gemm(const float* __restrict__ feat,
                                               const unsigned short* __restrict__ xs,
                                               float* __restrict__ out) {
    __shared__ v4u smemv[2 * PANEL_BYTES / 16];        // 64 KB, double-buffered A panels
    const char* smem = (const char*)smemv;
    const unsigned smem0 = (unsigned)(size_t)smemv;    // LDS byte offset (addr[31:0])

    const int tid  = threadIdx.x;
    const int wave = tid >> 5;
    const int lane = tid & 31;
    const int ncol = (blockIdx.x * 8 + wave) * 16 + (lane & 15);
    const int bkb  = (lane < 16) ? 0 : 8;
    const float* bbase = feat + (size_t)ncol * DDIM + bkb;
    const char*  xsb   = (const char*)xs;

    // B prologue: features K-step 0 (in flight while panel 0 stages)
    v4f f[2][4];
    {
        const v4f* bp = (const v4f*)bbase;
        f[0][0] = bp[0]; f[0][1] = bp[1]; f[0][2] = bp[4]; f[0][3] = bp[5];
    }
    // A prologue: async-stage panel 0
#pragma unroll
    for (int i = 0; i < 8; ++i) {
        const unsigned idx = (unsigned)(i * 256 + tid) * 16u;
        async16(smem0 + idx, xsb + idx);
    }
    asm volatile("s_wait_asynccnt 0x0" ::: "memory");
    __syncthreads();

    v8f acc[16];
    const v8f vz = {0.f,0.f,0.f,0.f,0.f,0.f,0.f,0.f};
#pragma unroll
    for (int m = 0; m < 16; ++m) acc[m] = vz;

    for (int p = 0; p < NPANELS; ++p) {
        // async-stage next A panel into the other LDS buffer
        if (p + 1 < NPANELS) {
            const char* src = xsb + (size_t)(p + 1) * PANEL_BYTES;
            const unsigned dst = smem0 + (unsigned)(((p + 1) & 1) * PANEL_BYTES);
#pragma unroll
            for (int i = 0; i < 8; ++i) {
                const unsigned idx = (unsigned)(i * 256 + tid) * 16u;
                async16(dst + idx, src + idx);
            }
        }
        const char* lp = smem + (p & 1) * PANEL_BYTES;

#pragma unroll
        for (int ksl = 0; ksl < PANEL_KS; ++ksl) {
            const int kg  = p * PANEL_KS + ksl;
            const int cur = kg & 1, nxt = cur ^ 1;

            // pack current B fragment (floats loaded one K-step ago)
            Frag bf;
            bf.u[0] = pk_bf16(f[cur][0][0], f[cur][0][1]);
            bf.u[1] = pk_bf16(f[cur][0][2], f[cur][0][3]);
            bf.u[2] = pk_bf16(f[cur][1][0], f[cur][1][1]);
            bf.u[3] = pk_bf16(f[cur][1][2], f[cur][1][3]);
            bf.u[4] = pk_bf16(f[cur][2][0], f[cur][2][1]);
            bf.u[5] = pk_bf16(f[cur][2][2], f[cur][2][3]);
            bf.u[6] = pk_bf16(f[cur][3][0], f[cur][3][1]);
            bf.u[7] = pk_bf16(f[cur][3][2], f[cur][3][3]);

            // issue next K-step's B loads (clamped at the tail) + deep prefetch
            {
                const int kn = (kg + 1 < KSTEPS) ? kg + 1 : kg;
                const v4f* bp = (const v4f*)(bbase + kn * 32);
                f[nxt][0] = bp[0]; f[nxt][1] = bp[1]; f[nxt][2] = bp[4]; f[nxt][3] = bp[5];
                const int kp = (kg + 6 < KSTEPS) ? kg + 6 : KSTEPS - 1;
                __builtin_prefetch(bbase + kp * 32, 0, 1);
            }

            // A fragments from LDS, ping-pong so ds latency hides behind a WMMA
            const char* fb = lp + ksl * (16 * 1024) + lane * 16;
            Frag a[2];
            a[0].q[0] = *(const v4u*)(fb);
            a[0].q[1] = *(const v4u*)(fb + 512);
#pragma unroll
            for (int m = 0; m < 16; ++m) {
                const int c = m & 1, n2 = c ^ 1;
                if (m + 1 < 16) {
                    a[n2].q[0] = *(const v4u*)(fb + (m + 1) * 1024);
                    a[n2].q[1] = *(const v4u*)(fb + (m + 1) * 1024 + 512);
                }
                acc[m] = __builtin_amdgcn_wmma_f32_16x16x32_bf16(
                    false, a[c].v, false, bf.v, (short)0, acc[m], false, false);
            }
        }
        asm volatile("s_wait_asynccnt 0x0" ::: "memory");
        __syncthreads();
    }

    // store C tiles: VGPR r -> M = r (lanes 0-15) / M = 8+r (lanes 16-31)
    const int rowoff = (lane < 16) ? 0 : 8;
#pragma unroll
    for (int m = 0; m < 16; ++m) {
#pragma unroll
        for (int r = 0; r < 8; ++r) {
            const int row = m * 16 + rowoff + r;
            out[(size_t)row * NDIM + ncol] = acc[m][r];
        }
    }
}

// ---------------- Kernel 3: per-row label-smoothed CE ----------------
__global__ __launch_bounds__(256) void cm_rowloss(const float* __restrict__ proto,
                                                  const float* __restrict__ memlog,
                                                  const int* __restrict__ targets,
                                                  float* __restrict__ rowloss) {
    const int b = blockIdx.x, tid = threadIdx.x;
    const float* prow = proto  + (size_t)b * PDIM;
    const float* mrow = memlog + (size_t)b * NDIM;
    __shared__ float red[256];

    float mx = -__FLT_MAX__;
    for (int j = tid; j < NCLS; j += 256) {
        float l = (j < PDIM ? prow[j] : mrow[j - PDIM]) * INV_TEMP;
        mx = fmaxf(mx, l);
    }
    red[tid] = mx; __syncthreads();
    for (int off = 128; off > 0; off >>= 1) {
        if (tid < off) red[tid] = fmaxf(red[tid], red[tid + off]);
        __syncthreads();
    }
    mx = red[0]; __syncthreads();

    float se = 0.f, sl = 0.f;
    for (int j = tid; j < NCLS; j += 256) {
        float l = (j < PDIM ? prow[j] : mrow[j - PDIM]) * INV_TEMP;
        se += expf(l - mx);
        sl += l;
    }
    red[tid] = se; __syncthreads();
    for (int off = 128; off > 0; off >>= 1) {
        if (tid < off) red[tid] += red[tid + off];
        __syncthreads();
    }
    se = red[0]; __syncthreads();
    red[tid] = sl; __syncthreads();
    for (int off = 128; off > 0; off >>= 1) {
        if (tid < off) red[tid] += red[tid + off];
        __syncthreads();
    }
    if (tid == 0) {
        sl = red[0];
        const float logZ   = mx + logf(se);
        const float lt     = mrow[targets[b]] * INV_TEMP;
        const float nll    = logZ - lt;
        const float smooth = logZ - sl * (1.0f / (float)NCLS);
        rowloss[b] = (1.0f - EPS_SM) * nll + EPS_SM * smooth;
    }
}

// ---------------- Kernel 4: mean over batch ----------------
__global__ __launch_bounds__(256) void cm_finalize(const float* __restrict__ rowloss,
                                                   float* __restrict__ out) {
    const int tid = threadIdx.x;
    __shared__ float red[256];
    red[tid] = rowloss[tid]; __syncthreads();
    for (int off = 128; off > 0; off >>= 1) {
        if (tid < off) red[tid] += red[tid + off];
        __syncthreads();
    }
    if (tid == 0) out[0] = red[0] * (1.0f / (float)BDIM);
}

extern "C" void kernel_launch(void* const* d_in, const int* in_sizes, int n_in,
                              void* d_out, int out_size, void* d_ws, size_t ws_size,
                              hipStream_t stream) {
    const float* inputs   = (const float*)d_in[0];   // [B, D]
    const int*   targets  = (const int*)d_in[1];     // [B]
    const float* proto    = (const float*)d_in[2];   // [B, P]
    const float* features = (const float*)d_in[3];   // [N, D]
    float* out = (float*)d_out;

    char* ws = (char*)d_ws;
    unsigned short* xs     = (unsigned short*)ws;                       // 1 MB, swizzled bf16 x
    float*          memlog = (float*)(ws + (1 << 20));                  // 64 MB logits
    float*          rloss  = (float*)(ws + (1 << 20) + (size_t)BDIM * NDIM * 4);

    cm_normalize<<<BDIM, 256, 0, stream>>>(inputs, xs);
    cm_gemm<<<NDIM / 128, 256, 0, stream>>>(features, xs, memlog);
    cm_rowloss<<<BDIM, 256, 0, stream>>>(proto, memlog, targets, rloss);
    cm_finalize<<<1, 256, 0, stream>>>(rloss, out);
}
